// MLA_4930622456428
// MI455X (gfx1250) — compile-verified, matrix operand order checked by reading, not run on hardware
//
#include <hip/hip_runtime.h>
#include <cstdint>
#include <cstddef>
#include <cmath>

// ---- problem dims ----
#define Hh   128
#define DN   128
#define DRr  64
#define DV   128
#define RQ   1536
#define RKV  512
#define HSZ  5120
#define SEQ  1024

typedef __attribute__((ext_vector_type(16))) __bf16 v16bf;
typedef __attribute__((ext_vector_type(8)))  __bf16 v8bf;
typedef __attribute__((ext_vector_type(8)))  float  v8f;

// ---------- helpers ----------
__device__ __forceinline__ __bf16 to_bf16(float f) {
    unsigned u = __builtin_bit_cast(unsigned, f);
    unsigned r = u + 0x7FFFu + ((u >> 16) & 1u);   // round-to-nearest-even
    unsigned short h = (unsigned short)(r >> 16);
    return __builtin_bit_cast(__bf16, h);
}

__device__ __forceinline__ v8bf load8bf(const __bf16* p) {
    return *reinterpret_cast<const v8bf*>(p);
}

__device__ __forceinline__ v16bf combine16(v8bf lo, v8bf hi) {
    return __builtin_shufflevector(lo, hi, 0,1,2,3,4,5,6,7,8,9,10,11,12,13,14,15);
}

__device__ __forceinline__ v8f wmma_bf16(v16bf a, v16bf b, v8f c) {
    return __builtin_amdgcn_wmma_f32_16x16x32_bf16(false, a, false, b, (short)0, c, false, false);
}

// A-fragment loader: row-major src, row stride ld (elements). Lane pattern per
// CDNA5 ISA 16-bit A 16x32 layout: lane<16 -> K {k..k+7, k+16..k+23},
// lane>=16 -> K {k+8..k+15, k+24..k+31}. (B uses the mirrored layout from Bt.)
__device__ __forceinline__ v16bf frag_rowmajor(const __bf16* base) {
    return combine16(load8bf(base), load8bf(base + 16));
}

// ---------- elementwise fp32 -> bf16 ----------
__global__ void cvt_bf16_kernel(const float* __restrict__ src, __bf16* __restrict__ dst, size_t n) {
    size_t i = (size_t)blockIdx.x * blockDim.x + threadIdx.x;
    if (i < n) dst[i] = to_bf16(src[i]);
}

// ---------- fp32 [rows][cols] -> bf16 transposed [cols][rows] ----------
__global__ void transpose_bf16_kernel(const float* __restrict__ src, __bf16* __restrict__ dst,
                                      int rows, int cols) {
    __shared__ float tile[32][33];
    int c0 = blockIdx.x * 32, r0 = blockIdx.y * 32;
    for (int i = threadIdx.y; i < 32; i += 8) {
        int r = r0 + i, c = c0 + threadIdx.x;
        tile[i][threadIdx.x] = (r < rows && c < cols) ? src[(size_t)r * cols + c] : 0.f;
    }
    __syncthreads();
    for (int i = threadIdx.y; i < 32; i += 8) {
        int c = c0 + i, r = r0 + threadIdx.x;
        if (c < cols && r < rows)
            dst[(size_t)c * rows + r] = to_bf16(tile[threadIdx.x][i]);
    }
}

// ---------- generic bf16 WMMA GEMM: C(MxN,f32) = A(MxK bf16 rowmajor) x Bt(NxK bf16 rowmajor)^T
// each wave computes a 16x64 tile of C. M%16==0, N%64==0, K%32==0.
__global__ void __launch_bounds__(256)
gemm_bf16_kernel(const __bf16* __restrict__ A, const __bf16* __restrict__ Bt,
                 float* __restrict__ C, int M, int N, int K) {
    int wave = blockIdx.x * (blockDim.x >> 5) + (threadIdx.x >> 5);
    int nt64 = N >> 6;
    int total = (M >> 4) * nt64;
    if (wave >= total) return;                 // wave-uniform: EXEC stays all-ones
    int mt = wave / nt64;
    int nt = wave % nt64;
    int lane = threadIdx.x & 31;
    int sub  = lane & 15;
    int half = lane >> 4;

    const __bf16* arow = A  + (size_t)(mt * 16 + sub) * K + half * 8;
    const __bf16* b0r  = Bt + (size_t)(nt * 64 + sub)      * K + half * 8;
    const __bf16* b1r  = b0r + (size_t)16 * K;
    const __bf16* b2r  = b0r + (size_t)32 * K;
    const __bf16* b3r  = b0r + (size_t)48 * K;

    v8f acc0 = {}, acc1 = {}, acc2 = {}, acc3 = {};
    for (int k = 0; k < K; k += 32) {
        __builtin_prefetch((const void*)(arow + k + 512), 0, 1);
        __builtin_prefetch((const void*)(b0r  + k + 512), 0, 1);
        v16bf af = frag_rowmajor(arow + k);
        v16bf bf0 = frag_rowmajor(b0r + k);
        v16bf bf1 = frag_rowmajor(b1r + k);
        v16bf bf2 = frag_rowmajor(b2r + k);
        v16bf bf3 = frag_rowmajor(b3r + k);
        acc0 = wmma_bf16(af, bf0, acc0);
        acc1 = wmma_bf16(af, bf1, acc1);
        acc2 = wmma_bf16(af, bf2, acc2);
        acc3 = wmma_bf16(af, bf3, acc3);
    }
    // C/D layout: lane l, vgpr r -> m = r + 8*(l>>4), n = l&15
    int m0 = mt * 16 + 8 * half;
    int n0 = nt * 64 + sub;
    float* crow;
#pragma unroll
    for (int r = 0; r < 8; ++r) {
        crow = C + (size_t)(m0 + r) * N;
        crow[n0 +  0] = acc0[r];
        crow[n0 + 16] = acc1[r];
        crow[n0 + 32] = acc2[r];
        crow[n0 + 48] = acc3[r];
    }
}

// ---------- RMS norm (fp32 in) -> bf16 out ----------
__global__ void rmsnorm_bf16_kernel(const float* __restrict__ src, int src_stride, int cols,
                                    const float* __restrict__ w,
                                    __bf16* __restrict__ dst, int dst_stride) {
    int row = blockIdx.x;
    const float* x = src + (size_t)row * src_stride;
    __shared__ float red[8];
    float s = 0.f;
    for (int c = threadIdx.x; c < cols; c += 256) { float v = x[c]; s += v * v; }
#pragma unroll
    for (int off = 16; off > 0; off >>= 1) s += __shfl_down(s, off);
    if ((threadIdx.x & 31) == 0) red[threadIdx.x >> 5] = s;
    __syncthreads();
    if (threadIdx.x == 0) {
        float t = 0.f;
#pragma unroll
        for (int i = 0; i < 8; ++i) t += red[i];
        red[0] = rsqrtf(t / (float)cols + 1e-6f);
    }
    __syncthreads();
    float rinv = red[0];
    __bf16* y = dst + (size_t)row * dst_stride;
    for (int c = threadIdx.x; c < cols; c += 256) y[c] = to_bf16(x[c] * rinv * w[c]);
}

// ---------- build qc[H][S][192] bf16 with RoPE on last 64 dims, pre-scaled ----------
__global__ void build_qc_kernel(const float* __restrict__ q, __bf16* __restrict__ qc, float scale) {
    int hs = blockIdx.x;
    int h = hs >> 10, s = hs & 1023;
    int d = threadIdx.x;                      // 0..191
    const float* row = q + (size_t)s * (Hh * 192) + (size_t)h * 192;
    float v;
    if (d < DN) {
        v = row[d];
    } else {
        int j = d - DN;                       // 0..63
        int i = j & 31;
        float ang = (float)s * __powf(10000.f, -(float)i / 32.f);
        float x = row[d];
        float other = (j < 32) ? -row[d + 32] : row[d - 32];
        v = x * __cosf(ang) + other * __sinf(ang);
    }
    qc[((size_t)h * SEQ + s) * 192 + d] = to_bf16(v * scale);
}

// ---------- build kc[H][S][192] bf16: k_nope from kv, roped k_rope broadcast ----------
__global__ void build_kc_kernel(const float* __restrict__ kv, const float* __restrict__ ckv_full,
                                __bf16* __restrict__ kc) {
    int hs = blockIdx.x;
    int h = hs >> 10, s = hs & 1023;
    int d = threadIdx.x;
    float v;
    if (d < DN) {
        v = kv[(size_t)s * (Hh * 256) + (size_t)h * 256 + d];
    } else {
        int j = d - DN;
        int i = j & 31;
        const float* kr = ckv_full + (size_t)s * (RKV + DRr) + RKV;
        float ang = (float)s * __powf(10000.f, -(float)i / 32.f);
        float x = kr[j];
        float other = (j < 32) ? -kr[j + 32] : kr[j - 32];
        v = x * __cosf(ang) + other * __sinf(ang);
    }
    kc[((size_t)h * SEQ + s) * 192 + d] = to_bf16(v);
}

// ---------- build v transposed: vt[H][128][1024] bf16 ----------
__global__ void build_vt_kernel(const float* __restrict__ kv, __bf16* __restrict__ vt) {
    size_t idx = (size_t)blockIdx.x * blockDim.x + threadIdx.x;   // ((h*128+d)*1024+s)
    int s = (int)(idx & 1023);
    size_t hd = idx >> 10;
    int d = (int)(hd & 127);
    int h = (int)(hd >> 7);
    vt[idx] = to_bf16(kv[(size_t)s * (Hh * 256) + (size_t)h * 256 + DN + d]);
}

// ---------- attention: one wave per (head, 16-query tile); full softmax over S=1024 ----------
__global__ void __launch_bounds__(32)
attn_kernel(const __bf16* __restrict__ qc, const __bf16* __restrict__ kc,
            const __bf16* __restrict__ vt, __bf16* __restrict__ attn_out) {
    __shared__ float sc[16 * 1024];           // 64KB: fp32 scores, later overlaid with bf16 P
    int h  = blockIdx.x >> 6;
    int q0 = (blockIdx.x & 63) * 16;
    int lane = threadIdx.x & 31;
    int sub  = lane & 15;
    int half = lane >> 4;

    const __bf16* qbase = qc + ((size_t)h * SEQ + q0) * 192;
    const __bf16* kbase = kc + (size_t)h * SEQ * 192;
    const __bf16* vbase = vt + (size_t)h * DV * SEQ;

    // Q fragments (K = 192 = 6 chunks of 32)
    v16bf qf[6];
#pragma unroll
    for (int i = 0; i < 6; ++i)
        qf[i] = frag_rowmajor(qbase + (size_t)sub * 192 + i * 32 + half * 8);

    // scores = (qc * scale) . kc
    for (int kt = 0; kt < 64; ++kt) {
        const __bf16* kb = kbase + (size_t)(kt * 16 + sub) * 192 + half * 8;
        v8f acc = {};
#pragma unroll
        for (int i = 0; i < 6; ++i)
            acc = wmma_bf16(qf[i], frag_rowmajor(kb + i * 32), acc);
#pragma unroll
        for (int r = 0; r < 8; ++r)
            sc[(r + 8 * half) * 1024 + kt * 16 + sub] = acc[r];
    }
    asm volatile("s_wait_dscnt 0" ::: "memory");

    // softmax per row; write bf16 P in place (byte 2k always <= already-read byte 4k)
    if (lane < 16) {
        float* row = sc + lane * 1024;
        float mx = -3.4e38f;
        for (int k = 0; k < 1024; ++k) mx = fmaxf(mx, row[k]);
        float sum = 0.f;
        for (int k = 0; k < 1024; ++k) sum += __expf(row[k] - mx);
        float inv = 1.0f / sum;
        __bf16* prow = (__bf16*)row;
        for (int k = 0; k < 1024; ++k) {
            float p = __expf(row[k] - mx) * inv;
            prow[k] = to_bf16(p);
        }
    }
    asm volatile("s_wait_dscnt 0" ::: "memory");

    // out = P (16x1024) @ V (1024x128), V supplied transposed
    const __bf16* P = (const __bf16*)sc;      // [16][1024] bf16
    v8f oacc[8];
#pragma unroll
    for (int nt = 0; nt < 8; ++nt) oacc[nt] = (v8f){};
    for (int kk = 0; kk < 1024; kk += 32) {
        v16bf af = frag_rowmajor(P + (size_t)sub * 1024 + kk + half * 8);
#pragma unroll
        for (int nt = 0; nt < 8; ++nt) {
            const __bf16* vb = vbase + (size_t)(nt * 16 + sub) * 1024 + kk + half * 8;
            oacc[nt] = wmma_bf16(af, frag_rowmajor(vb), oacc[nt]);
        }
    }
    // write attn_out [S][H*128] bf16
#pragma unroll
    for (int nt = 0; nt < 8; ++nt) {
#pragma unroll
        for (int r = 0; r < 8; ++r) {
            int m = r + 8 * half;
            int n = nt * 16 + sub;
            attn_out[(size_t)(q0 + m) * (Hh * DV) + (size_t)h * DV + n] = to_bf16(oacc[nt][r]);
        }
    }
}

// ---------- host launcher ----------
static inline size_t align256(size_t x) { return (x + 255) & ~(size_t)255; }

extern "C" void kernel_launch(void* const* d_in, const int* in_sizes, int n_in,
                              void* d_out, int out_size, void* d_ws, size_t ws_size,
                              hipStream_t stream) {
    (void)in_sizes; (void)n_in; (void)out_size; (void)ws_size;
    const float* hs        = (const float*)d_in[0];   // [1024][5120]
    const float* Wq_c      = (const float*)d_in[1];   // [5120][1536]
    const float* q_norm_w  = (const float*)d_in[2];   // [1536]
    const float* Wq_d      = (const float*)d_in[3];   // [1536][24576]
    const float* Wkv_c     = (const float*)d_in[4];   // [5120][576]
    const float* kv_norm_w = (const float*)d_in[5];   // [512]
    const float* Wkv_d     = (const float*)d_in[6];   // [512][32768]
    const float* Wo        = (const float*)d_in[7];   // [16384][5120]

    float* out_main = (float*)d_out;                           // [1024][5120]
    float* ckv_full = (float*)d_out + (size_t)SEQ * HSZ;       // [1024][576]

    char* ws = (char*)d_ws;
    size_t off = 0;
    auto alloc = [&](size_t bytes) { size_t o = off; off = align256(off + bytes); return o; };

    size_t o_hid_bf = alloc((size_t)SEQ * HSZ * 2);
    size_t o_Wqc_t  = alloc((size_t)RQ * HSZ * 2);
    size_t o_Wkvc_t = alloc((size_t)(RKV + DRr) * HSZ * 2);
    size_t o_Wqd_t  = alloc((size_t)(Hh * 192) * RQ * 2);
    size_t o_Wkvd_t = alloc((size_t)(Hh * 256) * RKV * 2);
    size_t o_Wo_t   = alloc((size_t)HSZ * (Hh * DV) * 2);
    size_t o_qc32   = alloc((size_t)SEQ * RQ * 4);
    size_t o_qcn    = alloc((size_t)SEQ * RQ * 2);
    size_t o_ckvn   = alloc((size_t)SEQ * RKV * 2);
    size_t o_q32    = alloc((size_t)SEQ * (Hh * 192) * 4);   // later reused for kc bf16
    size_t o_kv32   = alloc((size_t)SEQ * (Hh * 256) * 4);   // later reused for attn_out bf16
    size_t o_qcH    = alloc((size_t)Hh * SEQ * 192 * 2);
    size_t o_vt     = alloc((size_t)Hh * DV * SEQ * 2);

    __bf16* hid_bf = (__bf16*)(ws + o_hid_bf);
    __bf16* Wqc_t  = (__bf16*)(ws + o_Wqc_t);
    __bf16* Wkvc_t = (__bf16*)(ws + o_Wkvc_t);
    __bf16* Wqd_t  = (__bf16*)(ws + o_Wqd_t);
    __bf16* Wkvd_t = (__bf16*)(ws + o_Wkvd_t);
    __bf16* Wo_t   = (__bf16*)(ws + o_Wo_t);
    float*  qc32   = (float*)(ws + o_qc32);
    __bf16* qcn    = (__bf16*)(ws + o_qcn);
    __bf16* ckvn   = (__bf16*)(ws + o_ckvn);
    float*  q32    = (float*)(ws + o_q32);
    float*  kv32   = (float*)(ws + o_kv32);
    __bf16* qcH    = (__bf16*)(ws + o_qcH);
    __bf16* vt     = (__bf16*)(ws + o_vt);
    __bf16* kcH    = (__bf16*)(ws + o_q32);    // alias (q32 consumed by build_qc)
    __bf16* attn   = (__bf16*)(ws + o_kv32);   // alias (kv32 consumed by build_kc/vt)

    // 1) hidden -> bf16
    {
        size_t n = (size_t)SEQ * HSZ;
        cvt_bf16_kernel<<<(unsigned)((n + 255) / 256), 256, 0, stream>>>(hs, hid_bf, n);
    }
    // 2) transpose weights -> bf16 Bt
    dim3 tb(32, 8);
    transpose_bf16_kernel<<<dim3(RQ / 32, HSZ / 32), tb, 0, stream>>>(Wq_c, Wqc_t, HSZ, RQ);
    transpose_bf16_kernel<<<dim3((RKV + DRr) / 32, HSZ / 32), tb, 0, stream>>>(Wkv_c, Wkvc_t, HSZ, RKV + DRr);
    transpose_bf16_kernel<<<dim3((Hh * 192) / 32, RQ / 32), tb, 0, stream>>>(Wq_d, Wqd_t, RQ, Hh * 192);
    transpose_bf16_kernel<<<dim3((Hh * 256) / 32, RKV / 32), tb, 0, stream>>>(Wkv_d, Wkvd_t, RKV, Hh * 256);
    transpose_bf16_kernel<<<dim3(HSZ / 32, (Hh * DV) / 32), tb, 0, stream>>>(Wo, Wo_t, Hh * DV, HSZ);

    auto gemm = [&](const __bf16* A, const __bf16* Bt, float* C, int M, int N, int K) {
        int waves = (M / 16) * (N / 64);
        int blocks = (waves + 7) / 8;
        gemm_bf16_kernel<<<blocks, 256, 0, stream>>>(A, Bt, C, M, N, K);
    };

    // 3) q_c = hidden @ Wq_c ; ckv_full = hidden @ Wkv_c (written straight to output cache)
    gemm(hid_bf, Wqc_t, qc32, SEQ, RQ, HSZ);
    gemm(hid_bf, Wkvc_t, ckv_full, SEQ, RKV + DRr, HSZ);

    // 4) RMS norms -> bf16
    rmsnorm_bf16_kernel<<<SEQ, 256, 0, stream>>>(qc32, RQ, RQ, q_norm_w, qcn, RQ);
    rmsnorm_bf16_kernel<<<SEQ, 256, 0, stream>>>(ckv_full, RKV + DRr, RKV, kv_norm_w, ckvn, RKV);

    // 5) q = qcn @ Wq_d ; kv = ckvn @ Wkv_d
    gemm(qcn, Wqd_t, q32, SEQ, Hh * 192, RQ);
    gemm(ckvn, Wkvd_t, kv32, SEQ, Hh * 256, RKV);

    // 6) per-head layouts + RoPE (+ fold softmax scale into q)
    float scale = 1.0f / sqrtf((float)(DN + DRr));
    build_qc_kernel<<<Hh * SEQ, 192, 0, stream>>>(q32, qcH, scale);
    build_kc_kernel<<<Hh * SEQ, 192, 0, stream>>>(kv32, ckv_full, kcH);
    {
        size_t n = (size_t)Hh * DV * SEQ;
        build_vt_kernel<<<(unsigned)(n / 256), 256, 0, stream>>>(kv32, vt);
    }

    // 7) attention (1 wave per 16 queries per head)
    attn_kernel<<<Hh * (SEQ / 16), 32, 0, stream>>>(qcH, kcH, vt, attn);

    // 8) out = attn @ Wo
    gemm(attn, Wo_t, out_main, SEQ, HSZ, Hh * DV);
}
